// BinaryGroupConv_39444979646689
// MI455X (gfx1250) — 8-bit matrix operand order repaired and checked by reading, compiled, not run on hardware
//
#include <hip/hip_runtime.h>


typedef int v8i __attribute__((ext_vector_type(8)));

#define C_IN 256
#define NGRP 64
#define Hd   56
#define Wd   56
#define PW   58
#define EPSb 1e-5f

__global__ void prep_B_kernel(const float* __restrict__ wgt, int* __restrict__ Bws) {
  int g = blockIdx.x;
  int lane = threadIdx.x;
  int N = lane & 15;
  int khalf = lane >> 4;
  int dw[8];
#pragma unroll
  for (int v = 0; v < 8; ++v) {
    int t = (v < 4) ? (khalf * 4 + v) : (khalf * 4 + (v - 4) + 8);
    unsigned d = 0;
    if (N < 4 && t < 9) {
      int kh = t / 3, kw = t - (t / 3) * 3;
      const float* wp = wgt + (size_t)(g * 4 + N) * 36 + kh * 3 + kw;
#pragma unroll
      for (int ci = 0; ci < 4; ++ci) {
        float w = wp[ci * 9];
        unsigned b = (w > 0.f) ? 0x01u : 0xFFu;
        d |= (b & 0xFFu) << (8 * ci);
      }
    }
    dw[v] = (int)d;
  }
  int* o = Bws + (g * 32 + lane) * 8;
#pragma unroll
  for (int v = 0; v < 8; ++v) o[v] = dw[v];
}

__global__ __launch_bounds__(256) void binconv_kernel(
    const float* __restrict__ x, const float* __restrict__ gamma,
    const float* __restrict__ beta, const float* __restrict__ mean,
    const float* __restrict__ var, const int* __restrict__ Bws,
    float* __restrict__ out) {
  __shared__ int ldsA[PW * PW];
  __shared__ int dScr[8][64];

  int blk = blockIdx.x;
  int n = blk >> 6;
  int g = blk & 63;
  int tid = threadIdx.x;
  signed char* ldsAB = (signed char*)ldsA;

  for (int i = tid; i < PW * PW; i += 256) ldsA[i] = 0;
  __syncthreads();

  const float* xg = x + ((size_t)n * C_IN + (size_t)g * 4) * (Hd * Wd);
#pragma unroll
  for (int ci = 0; ci < 4; ++ci) {
    int c = g * 4 + ci;
    float inv = gamma[c] * rsqrtf(var[c] + EPSb);
    float bia = beta[c] - mean[c] * inv;
    for (int idx = tid; idx < Hd * Wd; idx += 256) {
      float v = xg[(size_t)ci * (Hd * Wd) + idx];
      int y = idx / Wd;
      int xx = idx - y * Wd;
      signed char s = (v * inv + bia) > 0.f ? (signed char)1 : (signed char)-1;
      ldsAB[(((y + 1) * PW + (xx + 1)) << 2) + ci] = s;
    }
  }
  __syncthreads();

  int wave = tid >> 5;
  int lane = tid & 31;
  int half = lane >> 4;
  int m = lane & 15;

  const int4* bp = (const int4*)(Bws + (g * 32 + lane) * 8);
  int4 b0 = bp[0], b1 = bp[1];
  v8i B;
  B[0] = b0.x; B[1] = b0.y; B[2] = b0.z; B[3] = b0.w;
  B[4] = b1.x; B[5] = b1.y; B[6] = b1.z; B[7] = b1.w;

  int offI[5], validv[5];
  const int cA0 = 0, cA1 = 1, cA2 = 2, cA3 = 3, cA4 = 8;
  int cAs[5] = {cA0, cA1, cA2, cA3, cA4};
#pragma unroll
  for (int v = 0; v < 5; ++v) {
    int t = cAs[v] + half * 4;
    validv[v] = (t < 9);
    int kh = t / 3, kw = t - (t / 3) * 3;
    offI[v] = validv[v] ? (kh * PW + kw) : 0;
  }

  for (int oy = wave; oy < Hd; oy += 8) {
#pragma unroll
    for (int tix = 0; tix < 4; ++tix) {
      int ox0 = (tix < 3) ? tix * 16 : 40;
      int base = oy * PW + (ox0 + m);

      v8i A;
#pragma unroll
      for (int v = 0; v < 5; ++v)
        A[v] = validv[v] ? ldsA[base + offI[v]] : 0;
      A[5] = 0; A[6] = 0; A[7] = 0;

      v8i Cz = {0, 0, 0, 0, 0, 0, 0, 0};
      v8i D = __builtin_amdgcn_wmma_i32_16x16x64_iu8(
          true, A, true, B, Cz, false, false);

      int co = lane & 15;
      if (co < 4) {
#pragma unroll
        for (int r = 0; r < 8; ++r)
          dScr[wave][co * 16 + half * 8 + r] = D[r];
      }
      __builtin_amdgcn_wave_barrier();
      asm volatile("s_wait_dscnt 0" ::: "memory");

#pragma unroll
      for (int h2 = 0; h2 < 2; ++h2) {
        int e = lane + h2 * 32;
        int col = e >> 4;
        int mm = e & 15;
        int cnew = col * 64 + g;
        size_t oidx = ((size_t)n * C_IN + cnew) * (Hd * Wd) +
                      (size_t)oy * Wd + (ox0 + mm);
        out[oidx] = (float)dScr[wave][e] + x[oidx];
      }
      __builtin_amdgcn_wave_barrier();
      asm volatile("s_wait_dscnt 0" ::: "memory");
    }
  }
}

extern "C" void kernel_launch(void* const* d_in, const int* in_sizes, int n_in,
                              void* d_out, int out_size, void* d_ws, size_t ws_size,
                              hipStream_t stream) {
  (void)in_sizes; (void)n_in; (void)out_size; (void)ws_size;
  const float* x     = (const float*)d_in[0];
  const float* wgt   = (const float*)d_in[1];
  const float* gamma = (const float*)d_in[2];
  const float* beta  = (const float*)d_in[3];
  const float* mean  = (const float*)d_in[4];
  const float* var   = (const float*)d_in[5];
  float* out = (float*)d_out;
  int* Bws = (int*)d_ws;

  prep_B_kernel<<<NGRP, 32, 0, stream>>>(wgt, Bws);
  binconv_kernel<<<32 * NGRP, 256, 0, stream>>>(x, gamma, beta, mean, var,
                                                Bws, out);
}
